// GINConvManual_352187319171
// MI455X (gfx1250) — matrix operand order repaired and checked by reading, not compiled, not run on hardware
//
#include <hip/hip_runtime.h>
#include <hip/hip_bf16.h>

#define N_NODES 40000
#define N_EDGES 640000
#define CH 128
#define BN_EPS 1e-5f
#define LDS_STRIDE 136   // 128 + 8 bf16 pad -> breaks 64-bank conflicts

typedef __attribute__((ext_vector_type(16))) __bf16 v16bf;
typedef __attribute__((ext_vector_type(8)))  float  v8f;

union FragAB {
    v16bf v;
    uint4 q[2];
};

// ---------------------------------------------------------------- init
__global__ void k_init(float* __restrict__ buf, float* __restrict__ sums) {
    const int total = N_NODES * CH;
    for (int i = blockIdx.x * blockDim.x + threadIdx.x; i < total;
         i += gridDim.x * blockDim.x)
        buf[i] = 0.0f;
    const int g = blockIdx.x * blockDim.x + threadIdx.x;
    if (g < 512) sums[g] = 0.0f;   // sum / sumsq / bnA / bnB
}

// ---------------------------------------------------------------- f32 -> bf16 weights
__global__ void k_cvt_w(const float* __restrict__ W1, const float* __restrict__ W2,
                        __bf16* __restrict__ w1b, __bf16* __restrict__ w2b) {
    const int i = blockIdx.x * blockDim.x + threadIdx.x;
    if (i < CH * CH) {
        w1b[i] = (__bf16)W1[i];
        w2b[i] = (__bf16)W2[i];
    }
}

// ---------------------------------------------------------------- edge scatter-add
// One wave per edge; lane handles 4 channels; f32 atomic adds (no return).
__global__ void k_scatter(const float* __restrict__ x,
                          const long long* __restrict__ ei,
                          float* __restrict__ agg) {
    const int lane   = threadIdx.x & 31;
    const int wave   = (blockIdx.x * blockDim.x + threadIdx.x) >> 5;
    const int nwaves = (gridDim.x * blockDim.x) >> 5;
    for (int e = wave; e < N_EDGES; e += nwaves) {
        const int src = (int)ei[e];
        const int dst = (int)ei[N_EDGES + e];
        const float4 v = ((const float4*)(x + (size_t)src * CH))[lane];
        float* a = agg + (size_t)dst * CH + lane * 4;
        atomicAdd(a + 0, v.x);
        atomicAdd(a + 1, v.y);
        atomicAdd(a + 2, v.z);
        atomicAdd(a + 3, v.w);
    }
}

// ---------------------------------------------------------------- GEMM1: h1 = ((1+eps)x + agg) @ W1^T + b1
// hbuf holds agg on entry, h1 on exit (in-place per 16-row block is safe:
// the whole tile is staged into LDS before any write).
__global__ void __launch_bounds__(256) k_gemm1(
        const float* __restrict__ x, float* __restrict__ hbuf,
        const __bf16* __restrict__ w1b, const float* __restrict__ b1,
        const float* __restrict__ eps_p,
        float* __restrict__ bsum, float* __restrict__ bsumsq) {
    __shared__ __bf16 As[16 * LDS_STRIDE];
    const int tid  = threadIdx.x;
    const int row0 = blockIdx.x * 16;
    const float se = 1.0f + eps_p[0];

    // stage 16x128 tile: h = (1+eps)*x + agg, converted to bf16
    {
        const int r = tid >> 4;
        const int c = (tid & 15) * 8;
        const float* xp = x    + (size_t)(row0 + r) * CH + c;
        const float* ap = hbuf + (size_t)(row0 + r) * CH + c;
        float4 x0 = *(const float4*)(xp);
        float4 x1 = *(const float4*)(xp + 4);
        float4 a0 = *(const float4*)(ap);
        float4 a1 = *(const float4*)(ap + 4);
        __bf16 t[8];
        t[0] = (__bf16)(se * x0.x + a0.x);
        t[1] = (__bf16)(se * x0.y + a0.y);
        t[2] = (__bf16)(se * x0.z + a0.z);
        t[3] = (__bf16)(se * x0.w + a0.w);
        t[4] = (__bf16)(se * x1.x + a1.x);
        t[5] = (__bf16)(se * x1.y + a1.y);
        t[6] = (__bf16)(se * x1.z + a1.z);
        t[7] = (__bf16)(se * x1.w + a1.w);
        *(uint4*)(As + r * LDS_STRIDE + c) = *(const uint4*)t;
    }
    __syncthreads();

    const int wv   = tid >> 5;     // N-tile (output-channel group of 16)
    const int lane = tid & 31;
    const int lid  = lane & 15;
    const int half = lane >> 4;

    v8f acc = {0.f, 0.f, 0.f, 0.f, 0.f, 0.f, 0.f, 0.f};
    const __bf16* wrow = w1b + (size_t)(wv * 16 + lid) * CH;
#pragma unroll
    for (int kc = 0; kc < 4; ++kc) {
        FragAB a, b;
        const __bf16* ap = As + lid * LDS_STRIDE + kc * 32 + half * 8;
        a.q[0] = *(const uint4*)(ap);
        a.q[1] = *(const uint4*)(ap + 16);
        const __bf16* bp = wrow + kc * 32 + half * 8;
        b.q[0] = *(const uint4*)(bp);
        b.q[1] = *(const uint4*)(bp + 16);
        acc = __builtin_amdgcn_wmma_f32_16x16x32_bf16(
                  false, a.v, false, b.v, (short)0, acc, false, false);
    }

    // writeback (C/D layout: VGPR r -> M = r + half*8, N = lid) + BN partials
    const int ch    = wv * 16 + lid;
    const float bias = b1[ch];
    float s = 0.f, s2 = 0.f;
#pragma unroll
    for (int r = 0; r < 8; ++r) {
        float v = acc[r] + bias;
        hbuf[(size_t)(row0 + half * 8 + r) * CH + ch] = v;
        s  += v;
        s2 += v * v;
    }
    s  += __shfl_down(s, 16);
    s2 += __shfl_down(s2, 16);
    if (half == 0) {
        atomicAdd(&bsum[ch], s);
        atomicAdd(&bsumsq[ch], s2);
    }
}

// ---------------------------------------------------------------- fold BN into scale/shift
__global__ void k_bn(const float* __restrict__ bsum, const float* __restrict__ bsumsq,
                     const float* __restrict__ gamma, const float* __restrict__ beta,
                     float* __restrict__ bnA, float* __restrict__ bnB) {
    const int ch = threadIdx.x;
    const float inv_n = 1.0f / (float)N_NODES;
    float mu  = bsum[ch] * inv_n;
    float var = bsumsq[ch] * inv_n - mu * mu;
    float rs  = rsqrtf(var + BN_EPS);
    float a   = rs * gamma[ch];
    bnA[ch] = a;
    bnB[ch] = beta[ch] - mu * a;
}

// ---------------------------------------------------------------- GEMM2: out = relu(BN(h1)) @ W2^T + b2 (in place)
__global__ void __launch_bounds__(256) k_gemm2(
        float* __restrict__ hbuf, const __bf16* __restrict__ w2b,
        const float* __restrict__ b2,
        const float* __restrict__ bnA, const float* __restrict__ bnB) {
    __shared__ __bf16 As[16 * LDS_STRIDE];
    const int tid  = threadIdx.x;
    const int row0 = blockIdx.x * 16;

    {
        const int r = tid >> 4;
        const int c = (tid & 15) * 8;
        const float* hp = hbuf + (size_t)(row0 + r) * CH + c;
        float4 h0 = *(const float4*)(hp);
        float4 h1 = *(const float4*)(hp + 4);
        float4 A0 = *(const float4*)(bnA + c);
        float4 A1 = *(const float4*)(bnA + c + 4);
        float4 B0 = *(const float4*)(bnB + c);
        float4 B1 = *(const float4*)(bnB + c + 4);
        __bf16 t[8];
        t[0] = (__bf16)fmaxf(fmaf(h0.x, A0.x, B0.x), 0.0f);
        t[1] = (__bf16)fmaxf(fmaf(h0.y, A0.y, B0.y), 0.0f);
        t[2] = (__bf16)fmaxf(fmaf(h0.z, A0.z, B0.z), 0.0f);
        t[3] = (__bf16)fmaxf(fmaf(h0.w, A0.w, B0.w), 0.0f);
        t[4] = (__bf16)fmaxf(fmaf(h1.x, A1.x, B1.x), 0.0f);
        t[5] = (__bf16)fmaxf(fmaf(h1.y, A1.y, B1.y), 0.0f);
        t[6] = (__bf16)fmaxf(fmaf(h1.z, A1.z, B1.z), 0.0f);
        t[7] = (__bf16)fmaxf(fmaf(h1.w, A1.w, B1.w), 0.0f);
        *(uint4*)(As + r * LDS_STRIDE + c) = *(const uint4*)t;
    }
    __syncthreads();

    const int wv   = tid >> 5;
    const int lane = tid & 31;
    const int lid  = lane & 15;
    const int half = lane >> 4;

    v8f acc = {0.f, 0.f, 0.f, 0.f, 0.f, 0.f, 0.f, 0.f};
    const __bf16* wrow = w2b + (size_t)(wv * 16 + lid) * CH;
#pragma unroll
    for (int kc = 0; kc < 4; ++kc) {
        FragAB a, b;
        const __bf16* ap = As + lid * LDS_STRIDE + kc * 32 + half * 8;
        a.q[0] = *(const uint4*)(ap);
        a.q[1] = *(const uint4*)(ap + 16);
        const __bf16* bp = wrow + kc * 32 + half * 8;
        b.q[0] = *(const uint4*)(bp);
        b.q[1] = *(const uint4*)(bp + 16);
        acc = __builtin_amdgcn_wmma_f32_16x16x32_bf16(
                  false, a.v, false, b.v, (short)0, acc, false, false);
    }

    const int ch    = wv * 16 + lid;
    const float bias = b2[ch];
#pragma unroll
    for (int r = 0; r < 8; ++r)
        hbuf[(size_t)(row0 + half * 8 + r) * CH + ch] = acc[r] + bias;
}

// ---------------------------------------------------------------- launch
extern "C" void kernel_launch(void* const* d_in, const int* in_sizes, int n_in,
                              void* d_out, int out_size, void* d_ws, size_t ws_size,
                              hipStream_t stream) {
    (void)in_sizes; (void)n_in; (void)out_size; (void)ws_size;
    const float*     x     = (const float*)d_in[0];
    const long long* ei    = (const long long*)d_in[1];
    const float*     W1    = (const float*)d_in[2];
    const float*     b1    = (const float*)d_in[3];
    const float*     gamma = (const float*)d_in[4];
    const float*     beta  = (const float*)d_in[5];
    const float*     W2    = (const float*)d_in[6];
    const float*     b2    = (const float*)d_in[7];
    const float*     eps   = (const float*)d_in[8];
    float* out = (float*)d_out;   // reused as agg -> h1 -> final output

    char*   ws    = (char*)d_ws;
    float*  bsum  = (float*)ws;          // 128
    float*  bsq   = bsum + 128;          // 128
    float*  bnA   = bsum + 256;          // 128
    float*  bnB   = bsum + 384;          // 128
    __bf16* w1b   = (__bf16*)(ws + 4096);        // 128*128 bf16
    __bf16* w2b   = w1b + CH * CH;               // 128*128 bf16

    k_init   <<<2048, 256, 0, stream>>>(out, bsum);
    k_cvt_w  <<<(CH * CH + 255) / 256, 256, 0, stream>>>(W1, W2, w1b, w2b);
    k_scatter<<<2048, 256, 0, stream>>>(x, ei, out);
    k_gemm1  <<<N_NODES / 16, 256, 0, stream>>>(x, out, w1b, b1, eps, bsum, bsq);
    k_bn     <<<1, 128, 0, stream>>>(bsum, bsq, gamma, beta, bnA, bnB);
    k_gemm2  <<<N_NODES / 16, 256, 0, stream>>>(out, w2b, b2, bnA, bnB);
}
